// GNN_drug_90563680403593
// MI455X (gfx1250) — matrix operand order repaired and checked by reading, not compiled, not run on hardware
//
#include <hip/hip_runtime.h>
#include <hip/hip_bf16.h>

// ---------------------------------------------------------------------------
// GNN (GCN x2 + GIN x3 + BN + segment-max readout) for MI455X (gfx1250).
// GEMMs use V_WMMA_F32_16X16X32_BF16 (wave32, 16x16 tiles, f32 accum).
// ---------------------------------------------------------------------------

typedef __bf16 bf16;
typedef __attribute__((ext_vector_type(16))) __bf16 v16bf;
typedef __attribute__((ext_vector_type(8)))  __bf16 v8bf;
typedef __attribute__((ext_vector_type(8)))  float  v8f;

#define N_NODES  100000
#define N_EDGES  400000
#define N_GRAPHS 2500
#define DIM      128
#define F_IN     78
#define KPAD     96        // F_IN padded to a multiple of 32 for WMMA K-steps
#define NPAD     100032    // N_NODES padded to a multiple of 64 (GEMM tile rows)
#define AGG_LD   80        // 78-dim aggregate buffer row stride (16B aligned)
#define BN_EPS   1e-5f
#define REP      (9 * DIM)

// ------------------------------- utility kernels ---------------------------

__global__ void k_zero(float* p, int n) {
  int i = blockIdx.x * blockDim.x + threadIdx.x;
  if (i < n) p[i] = 0.0f;
}

__global__ void k_copy(const float* __restrict__ in, float* __restrict__ out, int n) {
  int i = blockIdx.x * blockDim.x + threadIdx.x;
  if (i < n) out[i] = in[i];
}

// Cast f32 [in_rows x in_cols] (row stride in_ld) into bf16 [out_rows x out_ld],
// zero-filling column and row padding (so GEMM needs no bounds checks).
__global__ void k_cast_pad(const float* __restrict__ in, int in_rows, int in_ld,
                           int in_cols, bf16* __restrict__ out, int out_ld,
                           int out_rows) {
  int idx = blockIdx.x * blockDim.x + threadIdx.x;
  if (idx >= out_rows * out_ld) return;
  int r = idx / out_ld, c = idx - r * out_ld;
  float v = (r < in_rows && c < in_cols) ? in[(size_t)r * in_ld + c] : 0.0f;
  out[idx] = (bf16)v;
}

// W [K x 128] f32  ->  Wt [128 x kpad] bf16 (column-major-as-rows, K padded)
__global__ void k_transpose_w(const float* __restrict__ W, int K,
                              bf16* __restrict__ Wt, int kpad) {
  int idx = blockIdx.x * blockDim.x + threadIdx.x;
  if (idx >= DIM * kpad) return;
  int n = idx / kpad, k = idx - n * kpad;
  Wt[idx] = (bf16)((k < K) ? W[(size_t)k * DIM + n] : 0.0f);
}

// ------------------------------- degree / norm ------------------------------

__global__ void k_deg_init(float* deg) {
  int i = blockIdx.x * blockDim.x + threadIdx.x;
  if (i < NPAD) deg[i] = 1.0f;                    // self loop
}
__global__ void k_deg_count(const int* __restrict__ dst, float* deg) {
  int e = blockIdx.x * blockDim.x + threadIdx.x;
  if (e < N_EDGES) atomicAdd(&deg[dst[e]], 1.0f);
}
__global__ void k_dis(float* deg) {
  int i = blockIdx.x * blockDim.x + threadIdx.x;
  if (i < NPAD) deg[i] = rsqrtf(deg[i]);
}

// ------------------------------- WMMA GEMM ---------------------------------
// out[NPAD x 128] = A[NPAD x K](bf16) @ Wt^T  (+ bias) (+ relu)
// Wave -> 16x64 tile (4 accумulators share one A fragment).
// A-frag layout (ISA 16-bit A 16x32): lane<16: K = k..k+7 and k+16..k+23;
//                                     lane>=16: +8.   Two 16B loads.
// B-frag layout (matches A): lane<16: K = k..k+7 and k+16..k+23;
//                            lane>=16: +8.   Two 16B loads from Wt row.
__global__ __launch_bounds__(256) void k_wmma_gemm(
    const bf16* __restrict__ A, const bf16* __restrict__ Wt,
    const float* __restrict__ bias, float* __restrict__ out,
    int K, int flags /*1=bias, 2=relu*/) {
  int lane = threadIdx.x & 31;
  int wave = threadIdx.x >> 5;
  int rowbase = blockIdx.x * 64 + (wave >> 1) * 16;
  int colbase = (wave & 1) * 64;
  int m      = lane & 15;
  int khalfA = (lane >> 4) << 3;   // 0 or 8
  int khalfB = (lane >> 4) << 4;   // 0 or 16

  v8f acc[4] = {};
  const bf16* arow = A + (size_t)(rowbase + m) * K;

  for (int kk = 0; kk < K; kk += 32) {
    v16bf afrag;
    {
      const bf16* p = arow + kk + khalfA;
      v8bf lo = *(const v8bf*)p;
      v8bf hi = *(const v8bf*)(p + 16);
#pragma unroll
      for (int i = 0; i < 8; i++) { afrag[i] = lo[i]; afrag[8 + i] = hi[i]; }
    }
#pragma unroll
    for (int t = 0; t < 4; t++) {
      const bf16* q = Wt + (size_t)(colbase + t * 16 + m) * K + kk + khalfA;
      v8bf blo = *(const v8bf*)q;
      v8bf bhi = *(const v8bf*)(q + 16);
      v16bf bfrag;
#pragma unroll
      for (int i = 0; i < 8; i++) { bfrag[i] = blo[i]; bfrag[8 + i] = bhi[i]; }
      acc[t] = __builtin_amdgcn_wmma_f32_16x16x32_bf16(
          false, afrag, false, bfrag, (short)0, acc[t], false, false);
    }
  }

  int mrow = rowbase + khalfA;          // M = mrow + r  (r = VGPR index)
#pragma unroll
  for (int t = 0; t < 4; t++) {
    int col = colbase + t * 16 + m;
    float bv = (flags & 1) ? bias[col] : 0.0f;
#pragma unroll
    for (int r = 0; r < 8; r++) {
      float v = acc[t][r] + bv;
      if (flags & 2) v = fmaxf(v, 0.0f);
      out[(size_t)(mrow + r) * DIM + col] = v;
    }
  }
}

// ------------------------------- aggregation --------------------------------

// GCN self-loop init: out = lin * dis^2
__global__ void k_gcn_self(const float* __restrict__ lin,
                           const float* __restrict__ dis,
                           float* __restrict__ out) {
  int idx = blockIdx.x * blockDim.x + threadIdx.x;
  if (idx >= N_NODES * DIM) return;
  int i = idx >> 7;
  float d = dis[i];
  out[idx] = lin[idx] * d * d;
}

// GCN edge scatter: out[dst] += lin[src] * dis[src]*dis[dst]  (32 lanes/edge)
__global__ void k_gcn_edge(const int* __restrict__ src, const int* __restrict__ dst,
                           const float* __restrict__ dis,
                           const float* __restrict__ lin, float* __restrict__ out) {
  int t = blockIdx.x * blockDim.x + threadIdx.x;
  int e = t >> 5;
  if (e >= N_EDGES) return;
  int f = (t & 31) * 4;
  int s = src[e], d = dst[e];
  float w = dis[s] * dis[d];
  float4 v = *(const float4*)(lin + (size_t)s * DIM + f);
  float* o = out + (size_t)d * DIM + f;
  atomicAdd(o + 0, v.x * w);
  atomicAdd(o + 1, v.y * w);
  atomicAdd(o + 2, v.z * w);
  atomicAdd(o + 3, v.w * w);
}

// GIN edge scatter: out[dst*out_ld + f] += h[src*h_ld + f]   (32 lanes/edge)
__global__ void k_gin_edge(const int* __restrict__ src, const int* __restrict__ dst,
                           const float* __restrict__ h, int h_ld,
                           float* __restrict__ out, int out_ld, int F) {
  int t = blockIdx.x * blockDim.x + threadIdx.x;
  int e = t >> 5;
  if (e >= N_EDGES) return;
  int f = (t & 31) * 4;
  int s = src[e], d = dst[e];
#pragma unroll
  for (int j = 0; j < 4; j++) {
    int ff = f + j;
    if (ff < F) atomicAdd(&out[(size_t)d * out_ld + ff], h[(size_t)s * h_ld + ff]);
  }
}

// agg78 init: out[i*AGG_LD + c] = x[i*F_IN + c]  (zero pad cols)
__global__ void k_agg78_init(const float* __restrict__ x, float* __restrict__ out) {
  int idx = blockIdx.x * blockDim.x + threadIdx.x;
  if (idx >= N_NODES * AGG_LD) return;
  int i = idx / AGG_LD, c = idx - i * AGG_LD;
  out[idx] = (c < F_IN) ? x[(size_t)i * F_IN + c] : 0.0f;
}

__global__ void k_bias_relu(const float* __restrict__ in, const float* __restrict__ b,
                            float* __restrict__ out) {
  int idx = blockIdx.x * blockDim.x + threadIdx.x;
  if (idx >= N_NODES * DIM) return;
  out[idx] = fmaxf(in[idx] + b[idx & 127], 0.0f);
}

// ------------------------------- batch norm ---------------------------------

// out = relu(in); accumulate per-column sum / sumsq (blockDim = 128)
__global__ void k_relu_stats(const float* __restrict__ in, float* __restrict__ out,
                             float* __restrict__ sums) {
  int c = threadIdx.x;
  int row0 = blockIdx.x * 64;
  float s = 0.0f, q = 0.0f;
  for (int r = 0; r < 64; r++) {
    int i = row0 + r;
    if (i >= N_NODES) break;
    float v = fmaxf(in[(size_t)i * DIM + c], 0.0f);
    out[(size_t)i * DIM + c] = v;
    s += v; q += v * v;
  }
  atomicAdd(&sums[c], s);
  atomicAdd(&sums[DIM + c], q);
}

__global__ void k_bn_coef(const float* __restrict__ sums,
                          const float* __restrict__ gamma,
                          const float* __restrict__ beta,
                          float* __restrict__ coef) {
  int c = threadIdx.x;
  if (c >= DIM) return;
  float inv_n = 1.0f / (float)N_NODES;
  float mu  = sums[c] * inv_n;
  float var = sums[DIM + c] * inv_n - mu * mu;
  float a = gamma[c] * rsqrtf(var + BN_EPS);
  coef[c] = a;
  coef[DIM + c] = beta[c] - mu * a;
}

__global__ void k_bn_apply(float* __restrict__ x, const float* __restrict__ coef) {
  int idx = blockIdx.x * blockDim.x + threadIdx.x;
  if (idx >= N_NODES * DIM) return;
  int c = idx & 127;
  x[idx] = x[idx] * coef[c] + coef[DIM + c];
}

// ------------------------------- readout ------------------------------------

__device__ __forceinline__ int lower_bound_i(const int* a, int n, int key) {
  int lo = 0, hi = n;
  while (lo < hi) { int mid = (lo + hi) >> 1; if (a[mid] < key) lo = mid + 1; else hi = mid; }
  return lo;
}

// One block per graph: batch is sorted, binary search node range, max-reduce.
__global__ void k_pool(const float* __restrict__ xs0, const float* __restrict__ xs1,
                       const float* __restrict__ xs2, const float* __restrict__ g1,
                       const float* __restrict__ g2, const int* __restrict__ batch,
                       float* __restrict__ out) {
  int g  = blockIdx.x;
  int lo = lower_bound_i(batch, N_NODES, g);
  int hi = lower_bound_i(batch, N_NODES, g + 1);
  for (int f = threadIdx.x; f < REP; f += blockDim.x) {
    int chunk = f >> 7, c = f & 127;
    float acc = -__builtin_huge_valf();
    for (int i = lo; i < hi; i++) {
      size_t o = (size_t)i * DIM + c;
      float v;
      switch (chunk) {
        case 0: v = xs0[o]; break;
        case 1: v = xs1[o]; break;
        case 2: v = xs2[o]; break;
        case 3: v = xs0[o] * xs1[o] * xs2[o]; break;   // m
        case 4: v = xs0[o] + xs1[o] + xs2[o]; break;   // e
        case 5: v = g1[o]; break;
        case 6: v = g2[o]; break;
        case 7: v = g1[o] + g2[o]; break;              // g2a
        default: v = g1[o] * g2[o]; break;             // g2m
      }
      acc = fmaxf(acc, v);
    }
    out[(size_t)g * REP + f] = acc;
  }
}

// ------------------------------- driver -------------------------------------

extern "C" void kernel_launch(void* const* d_in, const int* in_sizes, int n_in,
                              void* d_out, int out_size, void* d_ws, size_t ws_size,
                              hipStream_t stream) {
  const float* x       = (const float*)d_in[0];
  const int*   ei      = (const int*)d_in[1];
  const int*   src     = ei;
  const int*   dst     = ei + N_EDGES;
  const int*   batch   = (const int*)d_in[2];
  const float* gcn1_W  = (const float*)d_in[4];
  const float* gcn1_b  = (const float*)d_in[5];
  const float* gcn2_W  = (const float*)d_in[6];
  const float* gcn2_b  = (const float*)d_in[7];
  const float* gin0_w1 = (const float*)d_in[8];
  const float* gin0_b1 = (const float*)d_in[9];
  const float* gin0_w2 = (const float*)d_in[10];
  const float* gin0_b2 = (const float*)d_in[11];
  const float* gin_w1  = (const float*)d_in[12];
  const float* gin_b1  = (const float*)d_in[13];
  const float* gin_w2  = (const float*)d_in[14];
  const float* gin_b2  = (const float*)d_in[15];
  const float* bn_g    = (const float*)d_in[16];
  const float* bn_b    = (const float*)d_in[17];
  float* out = (float*)d_out;

  // ---- workspace carve (~440 MB) ----
  char* ws = (char*)d_ws;
  size_t off = 0;
  auto carve = [&](size_t bytes) {
    void* p = ws + off;
    off += (bytes + 255) & ~(size_t)255;
    return p;
  };
  const size_t NB = (size_t)NPAD * DIM * sizeof(float);
  float* g1  = (float*)carve(NB);
  float* g2  = (float*)carve(NB);
  float* xs0 = (float*)carve(NB);
  float* xs1 = (float*)carve(NB);
  float* xs2 = (float*)carve(NB);
  float* T1  = (float*)carve(NB);
  float* T2  = (float*)carve(NB);
  bf16*  Abf = (bf16*)carve((size_t)NPAD * DIM * sizeof(bf16));
  bf16*  xb96 = (bf16*)carve((size_t)NPAD * KPAD * sizeof(bf16));
  float* agg78 = (float*)carve((size_t)NPAD * AGG_LD * sizeof(float));
  float* dis  = (float*)carve((size_t)NPAD * sizeof(float));
  bf16* wt_gcn1  = (bf16*)carve((size_t)DIM * KPAD * sizeof(bf16));
  bf16* wt_gcn2  = (bf16*)carve((size_t)DIM * DIM * sizeof(bf16));
  bf16* wt_g0w1  = (bf16*)carve((size_t)DIM * KPAD * sizeof(bf16));
  bf16* wt_g0w2  = (bf16*)carve((size_t)DIM * DIM * sizeof(bf16));
  bf16* wt_gw1[2], *wt_gw2[2];
  for (int l = 0; l < 2; l++) wt_gw1[l] = (bf16*)carve((size_t)DIM * DIM * sizeof(bf16));
  for (int l = 0; l < 2; l++) wt_gw2[l] = (bf16*)carve((size_t)DIM * DIM * sizeof(bf16));
  float* stats = (float*)carve(2 * DIM * sizeof(float));
  float* coef  = (float*)carve(2 * DIM * sizeof(float));

  const int TB = 256;
  const int gemm_grid = NPAD / 64;
  const int edge_grid = (N_EDGES * 32) / TB;
  const int nd_grid   = (N_NODES * DIM + TB - 1) / TB;
  const int cast128   = (NPAD * DIM + TB - 1) / TB;
  const int cast96    = (NPAD * KPAD + TB - 1) / TB;
  const int stat_grid = (N_NODES + 63) / 64;

  // ---- weight transposes (bf16, K padded) ----
  int wg96 = (DIM * KPAD + TB - 1) / TB, wg128 = (DIM * DIM + TB - 1) / TB;
  k_transpose_w<<<wg96, TB, 0, stream>>>(gcn1_W, F_IN, wt_gcn1, KPAD);
  k_transpose_w<<<wg128, TB, 0, stream>>>(gcn2_W, DIM, wt_gcn2, DIM);
  k_transpose_w<<<wg96, TB, 0, stream>>>(gin0_w1, F_IN, wt_g0w1, KPAD);
  k_transpose_w<<<wg128, TB, 0, stream>>>(gin0_w2, DIM, wt_g0w2, DIM);
  for (int l = 0; l < 2; l++) {
    k_transpose_w<<<wg128, TB, 0, stream>>>(gin_w1 + (size_t)l * DIM * DIM, DIM, wt_gw1[l], DIM);
    k_transpose_w<<<wg128, TB, 0, stream>>>(gin_w2 + (size_t)l * DIM * DIM, DIM, wt_gw2[l], DIM);
  }

  // ---- degree norm ----
  k_deg_init<<<(NPAD + TB - 1) / TB, TB, 0, stream>>>(dis);
  k_deg_count<<<(N_EDGES + TB - 1) / TB, TB, 0, stream>>>(dst, dis);
  k_dis<<<(NPAD + TB - 1) / TB, TB, 0, stream>>>(dis);

  // ---- GCN branch ----
  k_cast_pad<<<cast96, TB, 0, stream>>>(x, N_NODES, F_IN, F_IN, xb96, KPAD, NPAD);
  k_wmma_gemm<<<gemm_grid, TB, 0, stream>>>(xb96, wt_gcn1, nullptr, T1, KPAD, 0);
  k_gcn_self<<<nd_grid, TB, 0, stream>>>(T1, dis, T2);
  k_gcn_edge<<<edge_grid, TB, 0, stream>>>(src, dst, dis, T1, T2);
  k_bias_relu<<<nd_grid, TB, 0, stream>>>(T2, gcn1_b, g1);

  k_cast_pad<<<cast128, TB, 0, stream>>>(g1, N_NODES, DIM, DIM, Abf, DIM, NPAD);
  k_wmma_gemm<<<gemm_grid, TB, 0, stream>>>(Abf, wt_gcn2, nullptr, T1, DIM, 0);
  k_gcn_self<<<nd_grid, TB, 0, stream>>>(T1, dis, T2);
  k_gcn_edge<<<edge_grid, TB, 0, stream>>>(src, dst, dis, T1, T2);
  k_bias_relu<<<nd_grid, TB, 0, stream>>>(T2, gcn2_b, g2);

  // ---- GIN layer 0 (input x, K = 78) ----
  k_agg78_init<<<(N_NODES * AGG_LD + TB - 1) / TB, TB, 0, stream>>>(x, agg78);
  k_gin_edge<<<edge_grid, TB, 0, stream>>>(src, dst, x, F_IN, agg78, AGG_LD, F_IN);
  k_cast_pad<<<cast96, TB, 0, stream>>>(agg78, N_NODES, AGG_LD, F_IN, xb96, KPAD, NPAD);
  k_wmma_gemm<<<gemm_grid, TB, 0, stream>>>(xb96, wt_g0w1, gin0_b1, T1, KPAD, 3);
  k_cast_pad<<<cast128, TB, 0, stream>>>(T1, N_NODES, DIM, DIM, Abf, DIM, NPAD);
  k_wmma_gemm<<<gemm_grid, TB, 0, stream>>>(Abf, wt_g0w2, gin0_b2, T2, DIM, 1);
  k_zero<<<1, TB, 0, stream>>>(stats, 2 * DIM);
  k_relu_stats<<<stat_grid, DIM, 0, stream>>>(T2, xs0, stats);
  k_bn_coef<<<1, DIM, 0, stream>>>(stats, bn_g + 0 * DIM, bn_b + 0 * DIM, coef);
  k_bn_apply<<<nd_grid, TB, 0, stream>>>(xs0, coef);

  // ---- GIN layers 1..2 ----
  float* hs[3]  = { xs0, xs1, xs2 };
  for (int l = 0; l < 2; l++) {
    const float* h = hs[l];
    float* xo = hs[l + 1];
    k_copy<<<nd_grid, TB, 0, stream>>>(h, T2, N_NODES * DIM);
    k_gin_edge<<<edge_grid, TB, 0, stream>>>(src, dst, h, DIM, T2, DIM, DIM);
    k_cast_pad<<<cast128, TB, 0, stream>>>(T2, N_NODES, DIM, DIM, Abf, DIM, NPAD);
    k_wmma_gemm<<<gemm_grid, TB, 0, stream>>>(Abf, wt_gw1[l], gin_b1 + (size_t)l * DIM, T1, DIM, 3);
    k_cast_pad<<<cast128, TB, 0, stream>>>(T1, N_NODES, DIM, DIM, Abf, DIM, NPAD);
    k_wmma_gemm<<<gemm_grid, TB, 0, stream>>>(Abf, wt_gw2[l], gin_b2 + (size_t)l * DIM, T2, DIM, 1);
    k_zero<<<1, TB, 0, stream>>>(stats, 2 * DIM);
    k_relu_stats<<<stat_grid, DIM, 0, stream>>>(T2, xo, stats);
    k_bn_coef<<<1, DIM, 0, stream>>>(stats, bn_g + (size_t)(l + 1) * DIM, bn_b + (size_t)(l + 1) * DIM, coef);
    k_bn_apply<<<nd_grid, TB, 0, stream>>>(xo, coef);
  }

  // ---- readout: segment max over sorted batch ----
  k_pool<<<N_GRAPHS, TB, 0, stream>>>(xs0, xs1, xs2, g1, g2, batch, out);
}
